// SelfAttention_30322469110366
// MI455X (gfx1250) — matrix operand order repaired and checked by reading, not compiled, not run on hardware
//
#include <hip/hip_runtime.h>

// ---------------------------------------------------------------------------
// Self-attention for MI455X (gfx1250, wave32, WMMA f32_16x16x32_f16)
// B=64, S=512, D=768, H=12, HS=64
// Pipeline:
//   1) cvt_w_kernel  : Wq/Wk/Wv fp32 -> f16, transposed to [N][K]
//   2) qkv_kernel    : X @ W + b via WMMA (32Mx64N per wave); Q,K -> [B,H,S,HS],
//                      V -> [B,H,HS,S] (transposed), all f16
//   3) attn_kernel   : flash attention; K/V tiles double-buffered into LDS with
//                      global_load_async_to_lds_b128 + s_wait_asynccnt; scores
//                      computed transposed (K*Q^T) so the probability tile IS
//                      the WMMA A-fragment (no LDS transpose needed)
// ---------------------------------------------------------------------------

typedef __attribute__((ext_vector_type(16))) _Float16 v16h;
typedef __attribute__((ext_vector_type(8)))  _Float16 v8h;
typedef __attribute__((ext_vector_type(8)))  float    v8f;

constexpr int Bc  = 64;
constexpr int Sc  = 512;
constexpr int Dc  = 768;
constexpr int Hc  = 12;
constexpr int HSc = 64;
constexpr int Mtot = Bc * Sc;          // 32768 rows for the projections
constexpr int WSZ  = Dc * Dc;          // 589824 elements per weight matrix

__device__ __forceinline__ v16h frag2(v8h lo, v8h hi) {
  union { v8h h[2]; v16h v; } u;
  u.h[0] = lo; u.h[1] = hi;
  return u.v;
}

__device__ __forceinline__ v8h cvt8(v8f f) {
  v8h r;
#pragma unroll
  for (int i = 0; i < 8; ++i) r[i] = (_Float16)f[i];
  return r;
}

// Issue one 16-byte async global->LDS copy. Per ISA 10.2, the low 32 bits of a
// flat pointer into LDS are the LDS byte offset, which is what VDST wants.
__device__ __forceinline__ void async_copy16(const _Float16* gsrc, _Float16* ldst) {
  unsigned lds_off = (unsigned)(uintptr_t)ldst;
  unsigned long long gaddr = (unsigned long long)(uintptr_t)gsrc;
  asm volatile("global_load_async_to_lds_b128 %0, %1, off"
               :: "v"(lds_off), "v"(gaddr) : "memory");
}

// ------------------------- 1) weight convert + transpose --------------------
__global__ void cvt_w_kernel(const float* __restrict__ Wq,
                             const float* __restrict__ Wk,
                             const float* __restrict__ Wv,
                             _Float16* __restrict__ WT) {
  int idx = blockIdx.x * blockDim.x + threadIdx.x;   // 0 .. 3*D*D-1
  int z   = idx / WSZ;
  int rem = idx - z * WSZ;
  int k   = rem / Dc;
  int n   = rem - k * Dc;                            // n consecutive -> coalesced read
  const float* W = (z == 0) ? Wq : (z == 1 ? Wk : Wv);
  WT[z * WSZ + n * Dc + k] = (_Float16)W[k * Dc + n];
}

// ------------------------- 2) QKV projection GEMM ---------------------------
// grid = (Mtot/256, Dc/64, 3), block = 256 (8 waves); each wave: 32(M) x 64(N)
__global__ void qkv_kernel(const float* __restrict__ from,
                           const float* __restrict__ to,
                           const _Float16* __restrict__ WT,
                           const float* __restrict__ bq,
                           const float* __restrict__ bk,
                           const float* __restrict__ bv,
                           _Float16* __restrict__ Qf,
                           _Float16* __restrict__ Kf,
                           _Float16* __restrict__ VTf) {
  const int lane   = threadIdx.x & 31;
  const int wave   = threadIdx.x >> 5;
  const int lane16 = lane & 15;
  const int half   = lane >> 4;
  const int kb     = half * 16;              // 16-bit A-fragment K-base (contiguous layout)
  const int z      = blockIdx.z;
  const int mBase  = blockIdx.x * 256 + wave * 32;
  const int nBase  = blockIdx.y * 64;

  const float*    X    = (z == 0) ? from : to;
  const _Float16* W    = WT + z * WSZ;       // [N][K] row-major
  const float*    bias = (z == 0) ? bq : (z == 1 ? bk : bv);
  _Float16*       Y    = (z == 0) ? Qf : (z == 1 ? Kf : VTf);

  v8f acc[2][4];
#pragma unroll
  for (int u = 0; u < 2; ++u)
#pragma unroll
    for (int t = 0; t < 4; ++t)
#pragma unroll
      for (int i = 0; i < 8; ++i) acc[u][t][i] = 0.0f;

  const float* xrow0 = X + (size_t)(mBase + lane16) * Dc;
  const float* xrow1 = X + (size_t)(mBase + 16 + lane16) * Dc;

  for (int k0 = 0; k0 < Dc; k0 += 32) {
    // Two A fragments (rows mBase.. and mBase+16..), K pattern {kb..kb+15} (contiguous)
    v16h a[2];
    a[0] = frag2(cvt8(*(const v8f*)(xrow0 + k0 + kb)),
                 cvt8(*(const v8f*)(xrow0 + k0 + kb + 8)));
    a[1] = frag2(cvt8(*(const v8f*)(xrow1 + k0 + kb)),
                 cvt8(*(const v8f*)(xrow1 + k0 + kb + 8)));
#pragma unroll
    for (int t = 0; t < 4; ++t) {
      // B fragment: col = n, element i <-> K = k0 + half*16 + i (contiguous)
      const _Float16* bp = W + (size_t)(nBase + t * 16 + lane16) * Dc + k0 + half * 16;
      v16h b = frag2(*(const v8h*)bp, *(const v8h*)(bp + 8));
#pragma unroll
      for (int u = 0; u < 2; ++u)
        acc[u][t] = __builtin_amdgcn_wmma_f32_16x16x32_f16(
            false, a[u], false, b, (short)0, acc[u][t], false, false);
    }
  }

  // bias add + store. A 32-row wave tile never crosses a batch boundary
  // (32 | 512), so the batch index and row base are loop-invariant.
  const int bIdx = mBase >> 9;               // batch
  const int s0   = mBase & 511;              // sequence row base
  const int sLoc = 8 * half;                 // per-lane row offset within tile
#pragma unroll
  for (int t = 0; t < 4; ++t) {
    const int n    = nBase + t * 16 + lane16;
    const float bval = bias[n];
    const int h = n >> 6, d = n & 63;
    // Unified addressing: Q/K = [B,H,S,HS] (sStride=HS, d in base);
    //                     V^T = [B,H,HS,S] (sStride=1,  d*Sc in base).
    const size_t headBase = (size_t)(bIdx * Hc + h);
    const size_t base    = (z < 2) ? (headBase * Sc + s0) * HSc + d
                                   : (headBase * HSc + d) * Sc + s0;
    const int    sStride = (z < 2) ? HSc : 1;
#pragma unroll
    for (int u = 0; u < 2; ++u) {
#pragma unroll
      for (int r = 0; r < 8; ++r) {
        const int srel = u * 16 + r + sLoc;  // C layout: M = r + 8*half
        Y[base + (size_t)srel * sStride] = (_Float16)(acc[u][t][r] + bval);
      }
    }
  }
}

// ------------------------- 3) flash attention -------------------------------
// grid = (Sc/128, B*H), block = 256 (8 waves); each wave: 16 query rows.
// All 8 waves share the same (b,h) K/V tiles -> stage them in LDS with async
// copies, double-buffered. Scores computed TRANSPOSED (S^T = K * Q^T) so the
// post-softmax tile maps directly onto the PV A-fragment.
__global__ void attn_kernel(const _Float16* __restrict__ Qf,
                            const _Float16* __restrict__ Kf,
                            const _Float16* __restrict__ VTf,
                            float* __restrict__ out) {
  __shared__ alignas(16) _Float16 kbuf[2][32 * 64];   // [kk 32][d 64]
  __shared__ alignas(16) _Float16 vbuf[2][64 * 32];   // [d 64][kk 32]

  const int tid    = threadIdx.x;
  const int lane   = tid & 31;
  const int wave   = tid >> 5;
  const int lane16 = lane & 15;
  const int half   = lane >> 4;
  const int kb     = half * 8;
  const int bh     = blockIdx.y;
  const int qTile  = blockIdx.x * 128 + wave * 16;

  const _Float16* Qh = Qf  + (size_t)bh * Sc * HSc;
  const _Float16* Kh = Kf  + (size_t)bh * Sc * HSc;
  const _Float16* Vh = VTf + (size_t)bh * HSc * Sc;

  // stage K (contiguous 4KB) and V^T (64 rows x 32 cols) tiles for key block j0
  auto issue_stage = [&](int j0, int buf) {
    // K tile: rows j0..j0+31 are contiguous in [S,HS] layout
    async_copy16(Kh + (size_t)j0 * HSc + tid * 8, &kbuf[buf][tid * 8]);
    // V^T tile: row d, cols j0..j0+31 (stride Sc between rows)
    int d = tid >> 2, seg = (tid & 3) * 8;
    async_copy16(Vh + (size_t)d * Sc + j0 + seg, &vbuf[buf][d * 32 + seg]);
  };

  // Q as B-operand of S^T: col = q = qTile+lane16, K = d (contiguous halves)
  v16h bQ[2];
  {
    const _Float16* qp = Qh + (size_t)(qTile + lane16) * HSc;
#pragma unroll
    for (int j = 0; j < 2; ++j) {
      const _Float16* p = qp + j * 32 + half * 16;
      bQ[j] = frag2(*(const v8h*)p, *(const v8h*)(p + 8));
    }
  }

  v8f o[4];
#pragma unroll
  for (int t = 0; t < 4; ++t)
#pragma unroll
    for (int i = 0; i < 8; ++i) o[t][i] = 0.0f;

  float rowMax = -1e30f;
  float rowSum = 0.0f;
  const float LOG2E = 1.4426950408889634f;
  const float SCALE = 0.125f;                // 1/sqrt(HS)

  issue_stage(0, 0);

  constexpr int NIT = Sc / 32;               // 16 key blocks
  for (int it = 0; it < NIT; ++it) {
    const int cur = it & 1;
    if (it + 1 < NIT) {
      issue_stage((it + 1) * 32, cur ^ 1);
      // the 2 ops just issued may remain; older (stage it) ops must be done
      asm volatile("s_wait_asynccnt 0x2" ::: "memory");
    } else {
      asm volatile("s_wait_asynccnt 0x0" ::: "memory");
    }
    __syncthreads();                         // stage `it` visible to all waves

    // --- score tiles S^T[kk][q] from LDS K tile ---
    v8f sc[2];
#pragma unroll
    for (int t = 0; t < 2; ++t)
#pragma unroll
      for (int i = 0; i < 8; ++i) sc[t][i] = 0.0f;

#pragma unroll
    for (int t = 0; t < 2; ++t) {
      const _Float16* kp = &kbuf[cur][(t * 16 + lane16) * HSc];
#pragma unroll
      for (int j = 0; j < 2; ++j) {
        const _Float16* p = kp + j * 32 + half * 16;    // A layout: contiguous K halves
        v16h aK = frag2(*(const v8h*)p, *(const v8h*)(p + 8));
        sc[t] = __builtin_amdgcn_wmma_f32_16x16x32_f16(
            false, aK, false, bQ[j], (short)0, sc[t], false, false);
      }
    }

    // --- online softmax (per q = lane16, both halves kept in sync) ---
    float m = -1e30f;
#pragma unroll
    for (int t = 0; t < 2; ++t)
#pragma unroll
      for (int r = 0; r < 8; ++r) {
        sc[t][r] *= SCALE;
        m = fmaxf(m, sc[t][r]);
      }
    m = fmaxf(m, __shfl_xor(m, 16));
    float newMax = fmaxf(rowMax, m);
    float fscale = exp2f((rowMax - newMax) * LOG2E);

    float tsum = 0.0f;
#pragma unroll
    for (int t = 0; t < 2; ++t)
#pragma unroll
      for (int r = 0; r < 8; ++r) {
        float e = exp2f((sc[t][r] - newMax) * LOG2E);
        sc[t][r] = e;
        tsum += e;
      }
    tsum += __shfl_xor(tsum, 16);
    rowSum = rowSum * fscale + tsum;
    rowMax = newMax;

    // broadcast per-row rescale factor to the registers holding that row
    float fb[8];
#pragma unroll
    for (int r = 0; r < 8; ++r) fb[r] = __shfl(fscale, r + 8 * half);

    // P fragment == transposed score tiles, lane-local repack (the trick)
    v16h pf;
#pragma unroll
    for (int i = 0; i < 8; ++i) {
      pf[i]     = (_Float16)sc[0][i];
      pf[i + 8] = (_Float16)sc[1][i];
    }

    // --- PV from LDS V^T tile ---
#pragma unroll
    for (int t = 0; t < 4; ++t) {
#pragma unroll
      for (int r = 0; r < 8; ++r) o[t][r] *= fb[r];
      const _Float16* vp = &vbuf[cur][(t * 16 + lane16) * 32 + half * 8];
      v16h bV = frag2(*(const v8h*)vp, *(const v8h*)(vp + 16));
      o[t] = __builtin_amdgcn_wmma_f32_16x16x32_f16(
          false, pf, false, bV, (short)0, o[t], false, false);
    }

    __syncthreads();                         // done reading before next overwrite
  }

  // --- finalize: divide by rowSum, scatter to [B,S,D] fp32 ---
  float inv = 1.0f / rowSum;
  float li[8];
#pragma unroll
  for (int r = 0; r < 8; ++r) li[r] = __shfl(inv, r + 8 * half);

  const int b = bh / Hc, h = bh % Hc;
  float* orow = out + (size_t)(b * Sc + qTile) * Dc + h * 64;
#pragma unroll
  for (int t = 0; t < 4; ++t) {
    const int d = t * 16 + lane16;
#pragma unroll
    for (int r = 0; r < 8; ++r) {
      const int srel = r + 8 * half;
      orow[(size_t)srel * Dc + d] = o[t][r] * li[r];
    }
  }
}

// ---------------------------------------------------------------------------
extern "C" void kernel_launch(void* const* d_in, const int* in_sizes, int n_in,
                              void* d_out, int out_size, void* d_ws, size_t ws_size,
                              hipStream_t stream) {
  (void)in_sizes; (void)n_in; (void)out_size; (void)ws_size;

  const float* from = (const float*)d_in[0];
  const float* to   = (const float*)d_in[1];
  const float* Wq   = (const float*)d_in[2];
  const float* bq   = (const float*)d_in[3];
  const float* Wk   = (const float*)d_in[4];
  const float* bk   = (const float*)d_in[5];
  const float* Wv   = (const float*)d_in[6];
  const float* bv   = (const float*)d_in[7];
  float* out        = (float*)d_out;

  // workspace layout (f16): [WqT|WkT|WvT][Q][K][V^T]
  _Float16* ws  = (_Float16*)d_ws;
  _Float16* WT  = ws;                                    // 3 * 768*768
  _Float16* Qf  = WT + (size_t)3 * WSZ;                  // B*H*S*HS
  _Float16* Kf  = Qf + (size_t)Bc * Hc * Sc * HSc;
  _Float16* VTf = Kf + (size_t)Bc * Hc * Sc * HSc;

  cvt_w_kernel<<<dim3((3 * WSZ) / 256), dim3(256), 0, stream>>>(Wq, Wk, Wv, WT);

  qkv_kernel<<<dim3(Mtot / 256, Dc / 64, 3), dim3(256), 0, stream>>>(
      from, to, WT, bq, bk, bv, Qf, Kf, VTf);

  attn_kernel<<<dim3(Sc / 128, Bc * Hc), dim3(256), 0, stream>>>(Qf, Kf, VTf, out);
}
